// Bottleneck_37452114821417
// MI455X (gfx1250) — matrix operand order repaired and checked by reading, not compiled, not run on hardware
//
#include <hip/hip_runtime.h>
#include <hip/hip_bf16.h>

// CDNA5 / gfx1250 wave32 WMMA bottleneck kernel.
// h = relu(LN(feats @ W1)); h2 = relu(LN(gather(h) @ W2)); out = relu(LN(h2 @ W3) + feats)
//
// All matmuls run on v_wmma_f32_16x16x32_f16 (f16 in, f32 accumulate; every
// matmul is followed by LayerNorm which renormalizes, so f16 inputs are safe).
// Weights are pre-packed into per-lane B-fragment order and staged through LDS
// so the dominant conv2 loop reads B via ds_read_b128 with 8x block-level reuse.

typedef __attribute__((ext_vector_type(16))) _Float16 v16h;
typedef __attribute__((ext_vector_type(8)))  float    v8f;

#define LN_EPS 1e-6f

__device__ __forceinline__ v8f wmma_f16(v16h a, v16h b, v8f c) {
    // emits v_wmma_f32_16x16x32_f16
    return __builtin_amdgcn_wmma_f32_16x16x32_f16(
        /*neg_a=*/false, a, /*neg_b=*/false, b,
        /*c_mod=*/(short)0, c, /*reuse_a=*/false, /*reuse_b=*/false);
}

union AB16 { v16h h; uint4 q[2]; };

// A fragment (16x32 f16, MxK): lane<16 -> K runs [kb,kb+8) and [16+kb,16+kb+8),
// kb=0; lane>=16 -> kb=8.  Two 16B contiguous loads per lane.
__device__ __forceinline__ v16h load_a_f16(const _Float16* p32, int kbA) {
    AB16 u;
    u.q[0] = *reinterpret_cast<const uint4*>(p32 + kbA);
    u.q[1] = *reinterpret_cast<const uint4*>(p32 + 16 + kbA);
    return u.h;
}

// Same A fragment but source rows are f32 (feats): 4x float4 + convert.
__device__ __forceinline__ v16h load_a_f32(const float* p32, int kbA) {
    const float* r0 = p32 + kbA;
    const float* r1 = p32 + 16 + kbA;
    float4 a0 = *reinterpret_cast<const float4*>(r0);
    float4 a1 = *reinterpret_cast<const float4*>(r0 + 4);
    float4 a2 = *reinterpret_cast<const float4*>(r1);
    float4 a3 = *reinterpret_cast<const float4*>(r1 + 4);
    v16h h;
    h[0]  = (_Float16)a0.x; h[1]  = (_Float16)a0.y; h[2]  = (_Float16)a0.z; h[3]  = (_Float16)a0.w;
    h[4]  = (_Float16)a1.x; h[5]  = (_Float16)a1.y; h[6]  = (_Float16)a1.z; h[7]  = (_Float16)a1.w;
    h[8]  = (_Float16)a2.x; h[9]  = (_Float16)a2.y; h[10] = (_Float16)a2.z; h[11] = (_Float16)a2.w;
    h[12] = (_Float16)a3.x; h[13] = (_Float16)a3.y; h[14] = (_Float16)a3.z; h[15] = (_Float16)a3.w;
    return h;
}

// B fragment from LDS: lane reads 16 contiguous halfs (2x ds_read_b128).
__device__ __forceinline__ v16h load_b_lds(const _Float16* tile, int lane) {
    AB16 u;
    const uint4* p = reinterpret_cast<const uint4*>(tile + lane * 16);
    u.q[0] = p[0];
    u.q[1] = p[1];
    return u.h;
}

// LayerNorm(64) + ReLU + f16 store for one 16x64 accumulator group.
// acc VGPR layout: acc[t][r] = value at row (mbase + r + 8*hi), col (t*16 + rl).
__device__ __forceinline__ void ln64_relu_store(
    const v8f (&acc)[4], int mbase, int N, int rl, int hi,
    const float* __restrict__ g, const float* __restrict__ bb,
    _Float16* __restrict__ outp) {
#pragma unroll
    for (int r = 0; r < 8; ++r) {
        float s1 = 0.0f, s2 = 0.0f;
#pragma unroll
        for (int t = 0; t < 4; ++t) { float x = acc[t][r]; s1 += x; s2 += x * x; }
#pragma unroll
        for (int off = 1; off <= 8; off <<= 1) {
            s1 += __shfl_xor(s1, off);
            s2 += __shfl_xor(s2, off);
        }
        float mu   = s1 * (1.0f / 64.0f);
        float var  = s2 * (1.0f / 64.0f) - mu * mu;
        float rstd = rsqrtf(var + LN_EPS);
        int m = mbase + r + hi * 8;
        if (m < N) {
#pragma unroll
            for (int t = 0; t < 4; ++t) {
                int n = t * 16 + rl;
                float v = (acc[t][r] - mu) * rstd * g[n] + bb[n];
                v = fmaxf(v, 0.0f);
                outp[(size_t)m * 64 + n] = (_Float16)v;
            }
        }
    }
}

// ---------------------------------------------------------------------------
// Kernel 0: repack a [ksteps*32, ntiles*16] f32 row-major weight matrix into
// f16 B-fragment order: packed[tile][lane][j] = W[kk*32 + kb + (j<8 ? j : 8+j)][t*16 + (lane&15)],
// kb = (lane<16)?0:8, tile = kk*ntiles + t.
// ---------------------------------------------------------------------------
__global__ void repack_weights(const float* __restrict__ W, _Float16* __restrict__ Wp,
                               int ksteps, int ntiles) {
    int idx = blockIdx.x * blockDim.x + threadIdx.x;
    int total = ksteps * ntiles * 512;
    if (idx >= total) return;
    int tile = idx >> 9;
    int L    = (idx >> 4) & 31;
    int j    = idx & 15;
    int kk   = tile / ntiles;
    int t    = tile - kk * ntiles;
    int kb   = (L < 16) ? 0 : 8;
    int Nc   = ntiles * 16;
    int krow = kk * 32 + kb + (j < 8 ? j : 8 + j);
    int col  = t * 16 + (L & 15);
    Wp[idx] = (_Float16)W[(size_t)krow * Nc + col];
}

// ---------------------------------------------------------------------------
// Kernel 1: h = relu(LN(feats @ W1))  -> f16 [N,64]
// One wave = 16 points. K=256 -> 8 ksteps; Nc=64 -> 4 n-tiles.
// W1 (32 KB packed) staged in LDS once per block, shared by 4 waves.
// ---------------------------------------------------------------------------
__global__ __launch_bounds__(128) void conv1_ln_relu(
    const float* __restrict__ feats, const _Float16* __restrict__ w1p,
    const float* __restrict__ g1, const float* __restrict__ b1,
    _Float16* __restrict__ h, int N) {
    __shared__ _Float16 sW[8 * 4 * 512];   // 32 KB

    int tid  = threadIdx.x;
    int lane = tid & 31;
    int wid  = tid >> 5;
    int m0   = (blockIdx.x * 4 + wid) * 16;
    int rl   = lane & 15;
    int hi   = lane >> 4;
    int kbA  = hi ? 8 : 0;
    int rowA = m0 + rl; if (rowA >= N) rowA = N - 1;   // clamp; no early return (barrier)
    const float* frow = feats + (size_t)rowA * 256;

    {   // stage packed W1 -> LDS (2048 uint4, 16 per thread)
        const uint4* src = reinterpret_cast<const uint4*>(w1p);
        uint4* dst = reinterpret_cast<uint4*>(sW);
#pragma unroll
        for (int i = 0; i < 16; ++i) dst[tid + i * 128] = src[tid + i * 128];
    }
    __syncthreads();

    v8f acc[4];
#pragma unroll
    for (int t = 0; t < 4; ++t)
#pragma unroll
        for (int e = 0; e < 8; ++e) acc[t][e] = 0.0f;

#pragma unroll
    for (int kk = 0; kk < 8; ++kk) {
        v16h a = load_a_f32(frow + kk * 32, kbA);
#pragma unroll
        for (int t = 0; t < 4; ++t)
            acc[t] = wmma_f16(a, load_b_lds(sW + (kk * 4 + t) * 512, lane), acc[t]);
    }

    ln64_relu_store(acc, m0, N, rl, hi, g1, b1, h);
}

// ---------------------------------------------------------------------------
// Kernel 2: h2 = relu(LN(gather(h) @ W2)).  W2 is a 1728x64 B matrix
// (kc = k*64+c major).  One wave = 32 points (2 M-tiles) so each LDS-resident
// B fragment feeds 2 WMMAs; W2 staged through LDS in 24 KB chunks (3 k's),
// shared by 4 waves -> 8x reuse of every weight byte fetched from L2.
// ---------------------------------------------------------------------------
__global__ __launch_bounds__(128) void conv2_ln_relu(
    const _Float16* __restrict__ h, const int* __restrict__ nbr_idx,
    const _Float16* __restrict__ w2p,
    const float* __restrict__ g2, const float* __restrict__ b2,
    _Float16* __restrict__ h2, int N) {
    __shared__ _Float16 sW[3 * 2 * 4 * 512];   // 24 KB: 3 k's x 2 ksteps x 4 n-tiles

    int tid  = threadIdx.x;
    int lane = tid & 31;
    int wid  = tid >> 5;
    int m0   = (blockIdx.x * 4 + wid) * 32;    // 2 M-tiles of 16 points
    int rl   = lane & 15;
    int hi   = lane >> 4;
    int kbA  = hi ? 8 : 0;
    int ra0  = m0 + rl;      if (ra0 >= N) ra0 = N - 1;
    int ra1  = m0 + 16 + rl; if (ra1 >= N) ra1 = N - 1;
    const int* nbr0 = nbr_idx + (size_t)ra0 * 27;
    const int* nbr1 = nbr_idx + (size_t)ra1 * 27;

    v8f acc0[4], acc1[4];
#pragma unroll
    for (int t = 0; t < 4; ++t)
#pragma unroll
        for (int e = 0; e < 8; ++e) { acc0[t][e] = 0.0f; acc1[t][e] = 0.0f; }

    for (int c = 0; c < 9; ++c) {              // 9 chunks x 3 k's = 27
        __syncthreads();                       // previous chunk fully consumed
        {   // stage chunk c of packed W2 -> LDS (1536 uint4, 12 per thread)
            const uint4* src = reinterpret_cast<const uint4*>(w2p) + c * 1536;
            uint4* dst = reinterpret_cast<uint4*>(sW);
#pragma unroll
            for (int i = 0; i < 12; ++i) dst[tid + i * 128] = src[tid + i * 128];
        }
        __syncthreads();

#pragma unroll
        for (int kk = 0; kk < 3; ++kk) {
            int k = c * 3 + kk;
            const _Float16* h0 = h + (size_t)nbr0[k] * 64;
            const _Float16* h1 = h + (size_t)nbr1[k] * 64;
#pragma unroll
            for (int s = 0; s < 2; ++s) {
                v16h a0 = load_a_f16(h0 + s * 32, kbA);
                v16h a1 = load_a_f16(h1 + s * 32, kbA);
#pragma unroll
                for (int t = 0; t < 4; ++t) {
                    v16h b = load_b_lds(sW + ((kk * 2 + s) * 4 + t) * 512, lane);
                    acc0[t] = wmma_f16(a0, b, acc0[t]);
                    acc1[t] = wmma_f16(a1, b, acc1[t]);
                }
            }
        }
    }

    ln64_relu_store(acc0, m0,      N, rl, hi, g2, b2, h2);
    ln64_relu_store(acc1, m0 + 16, N, rl, hi, g2, b2, h2);
}

// ---------------------------------------------------------------------------
// Kernel 3: out = relu(LN(h2 @ W3) + feats).  K=64 -> 2 ksteps; Nc=256 -> 16
// n-tiles.  W3 (32 KB packed) staged in LDS once per block.
// ---------------------------------------------------------------------------
__global__ __launch_bounds__(128) void conv3_ln_res_relu(
    const _Float16* __restrict__ h2, const _Float16* __restrict__ w3p,
    const float* __restrict__ g3, const float* __restrict__ b3,
    const float* __restrict__ feats, float* __restrict__ out, int N) {
    __shared__ _Float16 sW[2 * 16 * 512];   // 32 KB

    int tid  = threadIdx.x;
    int lane = tid & 31;
    int wid  = tid >> 5;
    int m0   = (blockIdx.x * 4 + wid) * 16;
    int rl   = lane & 15;
    int hi   = lane >> 4;
    int kbA  = hi ? 8 : 0;
    int rowA = m0 + rl; if (rowA >= N) rowA = N - 1;
    const _Float16* hr = h2 + (size_t)rowA * 64;

    {   // stage packed W3 -> LDS
        const uint4* src = reinterpret_cast<const uint4*>(w3p);
        uint4* dst = reinterpret_cast<uint4*>(sW);
#pragma unroll
        for (int i = 0; i < 16; ++i) dst[tid + i * 128] = src[tid + i * 128];
    }
    __syncthreads();

    v8f acc[16];
#pragma unroll
    for (int t = 0; t < 16; ++t)
#pragma unroll
        for (int e = 0; e < 8; ++e) acc[t][e] = 0.0f;

#pragma unroll
    for (int s = 0; s < 2; ++s) {
        v16h a = load_a_f16(hr + s * 32, kbA);
#pragma unroll
        for (int t = 0; t < 16; ++t)
            acc[t] = wmma_f16(a, load_b_lds(sW + (s * 16 + t) * 512, lane), acc[t]);
    }

    // LayerNorm over 256 channels, residual add, relu, f32 store.
#pragma unroll
    for (int r = 0; r < 8; ++r) {
        float s1 = 0.0f, s2 = 0.0f;
#pragma unroll
        for (int t = 0; t < 16; ++t) { float x = acc[t][r]; s1 += x; s2 += x * x; }
#pragma unroll
        for (int off = 1; off <= 8; off <<= 1) {
            s1 += __shfl_xor(s1, off);
            s2 += __shfl_xor(s2, off);
        }
        float mu   = s1 * (1.0f / 256.0f);
        float var  = s2 * (1.0f / 256.0f) - mu * mu;
        float rstd = rsqrtf(var + LN_EPS);
        int m = m0 + r + hi * 8;
        if (m < N) {
            const float* frow = feats + (size_t)m * 256;
            float* orow = out + (size_t)m * 256;
#pragma unroll
            for (int t = 0; t < 16; ++t) {
                int n = t * 16 + rl;
                float v = (acc[t][r] - mu) * rstd * g3[n] + b3[n] + frow[n];
                orow[n] = fmaxf(v, 0.0f);
            }
        }
    }
}

// ---------------------------------------------------------------------------
extern "C" void kernel_launch(void* const* d_in, const int* in_sizes, int n_in,
                              void* d_out, int out_size, void* d_ws, size_t ws_size,
                              hipStream_t stream) {
    const float* feats = (const float*)d_in[0];
    const int*   nbr   = (const int*)d_in[1];
    const float* W1    = (const float*)d_in[2];
    const float* g1    = (const float*)d_in[3];
    const float* b1    = (const float*)d_in[4];
    const float* W2    = (const float*)d_in[5];
    const float* g2    = (const float*)d_in[6];
    const float* b2    = (const float*)d_in[7];
    const float* W3    = (const float*)d_in[8];
    const float* g3    = (const float*)d_in[9];
    const float* b3    = (const float*)d_in[10];
    float* out = (float*)d_out;

    const int N = in_sizes[0] / 256;   // C_IN = 256

    // workspace layout
    char* ws = (char*)d_ws;
    _Float16* w1p = (_Float16*)ws;                              //  8*4*512  halfs =  32 KB
    _Float16* w2p = (_Float16*)(ws + 32768);                    // 54*4*512  halfs = 216 KB
    _Float16* w3p = (_Float16*)(ws + 32768 + 221184);           //  2*16*512 halfs =  32 KB
    _Float16* h   = (_Float16*)(ws + 524288);                   // [N,64] f16
    _Float16* h2  = h + (size_t)N * 64;                         // [N,64] f16

    // repack weights into B-fragment order (f16)
    repack_weights<<<(8  * 4  * 512 + 255) / 256, 256, 0, stream>>>(W1, w1p, 8, 4);
    repack_weights<<<(54 * 4  * 512 + 255) / 256, 256, 0, stream>>>(W2, w2p, 54, 4);
    repack_weights<<<(2  * 16 * 512 + 255) / 256, 256, 0, stream>>>(W3, w3p, 2, 16);

    int waves16 = (N + 15) / 16;
    int blocks16 = (waves16 + 3) / 4;
    int waves32 = (N + 31) / 32;
    int blocks32 = (waves32 + 3) / 4;
    conv1_ln_relu    <<<blocks16, 128, 0, stream>>>(feats, w1p, g1, b1, h, N);
    conv2_ln_relu    <<<blocks32, 128, 0, stream>>>(h, nbr, w2p, g2, b2, h2, N);
    conv3_ln_res_relu<<<blocks16, 128, 0, stream>>>(h2, w3p, g3, b3, feats, out, N);
}
